// EncoderLayer_34205119545711
// MI455X (gfx1250) — matrix operand order repaired and checked by reading, not compiled, not run on hardware
//
#include <hip/hip_runtime.h>

// ---------------------------------------------------------------------------
// Types for CDNA5 WMMA (wave32, 16x16x32 bf16 -> f32)
// ---------------------------------------------------------------------------
typedef __attribute__((ext_vector_type(16))) __bf16 v16bf;
typedef __attribute__((ext_vector_type(8)))  float  v8f;

union Frag16 { v16bf v; unsigned int u[8]; };
union FragC  { v8f   v; float        f[8]; };

// ---- f32 -> bf16 (RNE); use HW packed convert when available --------------
#if __has_builtin(__builtin_amdgcn_cvt_pk_bf16_f32)
static __device__ __forceinline__ unsigned int pack2(float a, float b) {
    auto r = __builtin_amdgcn_cvt_pk_bf16_f32(a, b);
    return __builtin_bit_cast(unsigned int, r);
}
static __device__ __forceinline__ unsigned short f2b(float f) {
    return (unsigned short)pack2(f, 0.0f);
}
#else
static __device__ __forceinline__ unsigned short f2b(float f) {
    unsigned int u = __float_as_uint(f);
    unsigned int r = (u + 0x7FFFu + ((u >> 16) & 1u)) >> 16;   // RNE
    return (unsigned short)r;
}
static __device__ __forceinline__ unsigned int pack2(float a, float b) {
    return (unsigned int)f2b(a) | ((unsigned int)f2b(b) << 16);
}
#endif

// ---- width-16 xor-butterfly via DPP16 (no LDS traffic) --------------------
template <int CTRL>
static __device__ __forceinline__ float dpp_f(float x) {
    return __int_as_float(__builtin_amdgcn_mov_dpp(__float_as_int(x), CTRL, 0xF, 0xF, true));
}
static __device__ __forceinline__ float red16_max(float x) {
    x = fmaxf(x, dpp_f<0xB1>(x));    // quad_perm [1,0,3,2]  (xor1)
    x = fmaxf(x, dpp_f<0x4E>(x));    // quad_perm [2,3,0,1]  (xor2)
    x = fmaxf(x, dpp_f<0x141>(x));   // row_half_mirror      (xor7)
    x = fmaxf(x, dpp_f<0x140>(x));   // row_mirror           (xor15)
    return x;
}
static __device__ __forceinline__ float red16_sum(float x) {
    x += dpp_f<0xB1>(x);
    x += dpp_f<0x4E>(x);
    x += dpp_f<0x141>(x);
    x += dpp_f<0x140>(x);
    return x;
}

// ---- async global->LDS copies (ASYNCcnt-tracked) --------------------------
#define ASYNC_COPY 1
#if ASYNC_COPY
static __device__ __forceinline__ void async_b128(void* lds, const void* g) {
    asm volatile("global_load_async_to_lds_b128 %0, %1, off"
                 :: "v"((unsigned int)(unsigned long long)lds),
                    "v"((unsigned long long)g)
                 : "memory");
}
static __device__ __forceinline__ void wait_async0() {
    asm volatile("s_wait_asynccnt 0" ::: "memory");
}
#endif

// ---------------------------------------------------------------------------
// WMMA GEMM: C[M,N] = act(A[M,K] @ Bt^T + bias).
//   Bt is PRE-TRANSPOSED weights, bf16, [N][K] row-major -> contiguous
//   staging via async b128 global->LDS copies.
//   Double-buffered LDS pipeline: per K-step, load fragments of the current
//   tile, issue async copies for the next tile, run WMMAs while the async
//   engine fills the other buffer, then s_wait_asynccnt + one barrier.
// Tile BM=128, BN=64, BK=32; 256 thr = 8 waves (4x2), each wave 32x32.
// All dims here are multiples of 128/64 -> no bounds checks.
// ---------------------------------------------------------------------------
#define BM 128
#define BN 64
#define BKK 32
#define APITCH 48   // u16 units; 96B rows (16B-aligned for b128)
#define BPITCH 48

enum { EPI_NONE = 0, EPI_BIAS = 1, EPI_GELU = 2, EPI_GATE = 3 };

template <bool A_IS_F32, bool OUT_BF16, bool OUT_TRANS, int EPI>
__global__ __launch_bounds__(256) void gemm_wmma(
    const void* __restrict__ Av, const unsigned short* __restrict__ Bt,
    const float* __restrict__ bias, const float* __restrict__ aux,
    void* __restrict__ Outv, int M, int N, int K)
{
    __shared__ unsigned short sA[2][BM * APITCH];
    __shared__ unsigned short sBT[2][BN * BPITCH];

    const int t    = threadIdx.x;
    const int wid  = t >> 5;
    const int lane = t & 31;
    const int m0   = blockIdx.x * BM;
    const int n0   = blockIdx.y * BN;
    const int wm   = (wid >> 1) * 32;       // 0,32,64,96
    const int wn   = (wid & 1) * 32;        // 0,32

    const float*          Af = (const float*)Av;
    const unsigned short* Ab = (const unsigned short*)Av;

    // ---- tile staging ------------------------------------------------------
    auto stage = [&](int buf, int kt) {
        if (A_IS_F32) {
            // 128x32 f32 -> bf16 : 2048 pairs, 8 per thread
#pragma unroll
            for (int j = 0; j < 8; ++j) {
                int p   = t + j * 256;
                int row = p >> 4;
                int cp  = p & 15;
                float2 f2 = *(const float2*)&Af[(size_t)(m0 + row) * K + kt + 2 * cp];
                *(unsigned int*)&sA[buf][row * APITCH + 2 * cp] = pack2(f2.x, f2.y);
            }
        } else {
            // 128x32 bf16 : 512 x 16B chunks, 2 per thread
#pragma unroll
            for (int j = 0; j < 2; ++j) {
                int p   = t + j * 256;
                int row = p >> 2;
                int c16 = (p & 3) * 8;
#if ASYNC_COPY
                async_b128(&sA[buf][row * APITCH + c16],
                           &Ab[(size_t)(m0 + row) * K + kt + c16]);
#else
                *(uint4*)&sA[buf][row * APITCH + c16] =
                    *(const uint4*)&Ab[(size_t)(m0 + row) * K + kt + c16];
#endif
            }
        }
        // 64x32 bf16 B tile : 256 x 16B chunks, 1 per thread
        {
            int n   = t >> 2;
            int c16 = (t & 3) * 8;
#if ASYNC_COPY
            async_b128(&sBT[buf][n * BPITCH + c16],
                       &Bt[(size_t)(n0 + n) * K + kt + c16]);
#else
            *(uint4*)&sBT[buf][n * BPITCH + c16] =
                *(const uint4*)&Bt[(size_t)(n0 + n) * K + kt + c16];
#endif
        }
    };

    FragC acc[2][2];
#pragma unroll
    for (int i = 0; i < 2; ++i)
#pragma unroll
        for (int j = 0; j < 2; ++j)
#pragma unroll
            for (int r = 0; r < 8; ++r) acc[i][j].f[r] = 0.0f;

    // prologue: fill buffer 0
    stage(0, 0);
#if ASYNC_COPY
    wait_async0();
#endif
    __syncthreads();

    const int koffA = (lane >> 4) << 3;   // 0 or 8
    const int koffB = (lane >> 4) << 4;   // 0 or 16

    for (int kt = 0, it = 0; kt < K; kt += BKK, ++it) {
        const int cur = it & 1;

        // ---- load current tile's fragments FIRST (keeps the in-order DS
        //      pipeline from serializing compute behind next-tile staging) --
        Frag16 a[2], b[2];
#pragma unroll
        for (int i = 0; i < 2; ++i) {
            int mrow = wm + i * 16 + (lane & 15);
#pragma unroll
            for (int g = 0; g < 2; ++g)
#pragma unroll
                for (int vi = 0; vi < 4; ++vi)
                    a[i].u[g * 4 + vi] =
                        *(const unsigned int*)&sA[cur][mrow * APITCH + g * 16 + koffA + 2 * vi];
        }
#pragma unroll
        for (int j = 0; j < 2; ++j) {
            int ncol = wn + j * 16 + (lane & 15);
#pragma unroll
            for (int v = 0; v < 8; ++v)
                b[j].u[v] = *(const unsigned int*)&sBT[cur][ncol * BPITCH + (v >> 2) * 16 + koffA + 2 * (v & 3)];
        }

        // ---- issue staging for the next tile into the other buffer --------
        if (kt + BKK < K) stage(cur ^ 1, kt + BKK);

        // ---- WMMAs overlap the async copies -------------------------------
#pragma unroll
        for (int i = 0; i < 2; ++i)
#pragma unroll
            for (int j = 0; j < 2; ++j)
                acc[i][j].v = __builtin_amdgcn_wmma_f32_16x16x32_bf16(
                    false, a[i].v, false, b[j].v, (short)0, acc[i][j].v, false, false);

#if ASYNC_COPY
        wait_async0();
#endif
        __syncthreads();
    }

    // ---- epilogue ---------------------------------------------------------
#pragma unroll
    for (int i = 0; i < 2; ++i)
#pragma unroll
        for (int j = 0; j < 2; ++j) {
            int mbase = m0 + wm + i * 16 + ((lane >> 4) << 3);
            int n     = n0 + wn + j * 16 + (lane & 15);
            float bv  = (EPI != EPI_NONE) ? bias[n] : 0.0f;
#pragma unroll
            for (int r = 0; r < 8; ++r) {
                int    m   = mbase + r;
                size_t off = OUT_TRANS ? ((size_t)n * M + m) : ((size_t)m * N + n);
                float  v   = acc[i][j].f[r] + bv;
                if (EPI == EPI_GELU) v = 0.5f * v * (1.0f + erff(v * 0.70710678118654752f));
                if (EPI == EPI_GATE) {
                    float g = aux[off];
                    v = v * (1.0f / (1.0f + __expf(-g)));
                }
                if (OUT_BF16) ((unsigned short*)Outv)[off] = f2b(v);
                else          ((float*)Outv)[off]          = v;
            }
        }
}

// ---------------------------------------------------------------------------
// Flash attention, q == k == v (bf16 in), head dim 64, scale 1/sqrt(E).
// Block: 128 threads (4 waves) -> 64 query rows of one (b,h) pair.
// ---------------------------------------------------------------------------
__global__ __launch_bounds__(128) void attn_flash(
    const unsigned short* __restrict__ qb, float* __restrict__ out,
    int Bn, int L, int E, int H)
{
    const int dh = 64;
    __shared__ unsigned short sKV[32 * 72];       // [key][d]   (== B^T for S)
    __shared__ unsigned short sVT[64 * 40];       // [d][key]   (== B^T for P@V)
    __shared__ unsigned short sP[4 * 16 * 40];    // per-wave P tile 16x32

    const int bh    = blockIdx.x;
    const int b     = bh >> 3;
    const int h     = bh & 7;
    const int qbase = blockIdx.y * 64;
    const int t     = threadIdx.x;
    const int wid   = t >> 5;
    const int lane  = t & 31;
    const int half8 = (lane >> 4) << 3;           // C-layout row offset
    const int rowb  = wid * 16;                   // wave's rows inside sP

    // ---- Q fragments (2 K-chunks of 32 dims), direct bf16 global loads ----
    Frag16 qa[2];
    {
        int m    = lane & 15;
        int qr   = qbase + wid * 16 + m;
        int koff = (lane >> 4) << 3;
        const unsigned short* qrow = qb + ((size_t)(b * L + qr)) * E + h * dh;
#pragma unroll
        for (int c = 0; c < 2; ++c)
#pragma unroll
            for (int g = 0; g < 2; ++g)
#pragma unroll
                for (int vi = 0; vi < 4; ++vi) {
                    int k = c * 32 + g * 16 + koff + 2 * vi;
                    qa[c].u[g * 4 + vi] = *(const unsigned int*)&qrow[k];
                }
    }

    float mr[8], lr[8];
    FragC o[4];
#pragma unroll
    for (int r = 0; r < 8; ++r) { mr[r] = -1e30f; lr[r] = 0.0f; }
#pragma unroll
    for (int dt = 0; dt < 4; ++dt)
#pragma unroll
        for (int r = 0; r < 8; ++r) o[dt].f[r] = 0.0f;

    const float scale = 0.0441941738241592f;      // 1/sqrt(512)

    for (int kv = 0; kv < L; kv += 32) {
        __syncthreads();
        // ---- stage 32 keys x 64 dims (row-major + transposed) -------------
#pragma unroll
        for (int j = 0; j < 4; ++j) {
            int p   = t + j * 128;                // 512 quads of 4 bf16
            int key = p >> 4;
            int d0  = (p & 15) * 4;
            unsigned long long v4 =
                *(const unsigned long long*)&qb[((size_t)(b * L + kv + key)) * E + h * dh + d0];
            *(unsigned long long*)&sKV[key * 72 + d0] = v4;
            sVT[(d0 + 0) * 40 + key] = (unsigned short)v4;
            sVT[(d0 + 1) * 40 + key] = (unsigned short)(v4 >> 16);
            sVT[(d0 + 2) * 40 + key] = (unsigned short)(v4 >> 32);
            sVT[(d0 + 3) * 40 + key] = (unsigned short)(v4 >> 48);
        }
        __syncthreads();

        // ---- S tiles: 16 queries x 32 keys --------------------------------
        FragC s[2];
        const int koffB = (lane >> 4) << 4;
#pragma unroll
        for (int nt = 0; nt < 2; ++nt) {
#pragma unroll
            for (int r = 0; r < 8; ++r) s[nt].f[r] = 0.0f;
            int key = nt * 16 + (lane & 15);
#pragma unroll
            for (int c = 0; c < 2; ++c) {
                Frag16 kb;
#pragma unroll
                for (int v = 0; v < 8; ++v)
                    kb.u[v] = *(const unsigned int*)&sKV[key * 72 + c * 32 + (v >> 2) * 16 + (koffB >> 1) + 2 * (v & 3)];
                s[nt].v = __builtin_amdgcn_wmma_f32_16x16x32_bf16(
                    false, qa[c].v, false, kb.v, (short)0, s[nt].v, false, false);
            }
        }

        // ---- online softmax (DPP reductions) ------------------------------
#pragma unroll
        for (int r = 0; r < 8; ++r) {
            float v0 = s[0].f[r] * scale;
            float v1 = s[1].f[r] * scale;
            float rm = red16_max(fmaxf(v0, v1));
            float mn   = fmaxf(mr[r], rm);
            float corr = __expf(mr[r] - mn);
            float p0   = __expf(v0 - mn);
            float p1   = __expf(v1 - mn);
            float rs   = red16_sum(p0 + p1);
            lr[r] = lr[r] * corr + rs;
            mr[r] = mn;
#pragma unroll
            for (int dt = 0; dt < 4; ++dt) o[dt].f[r] *= corr;
            int rl = r + half8;
            sP[(rowb + rl) * 40 + (lane & 15)]      = f2b(p0);
            sP[(rowb + rl) * 40 + 16 + (lane & 15)] = f2b(p1);
        }

        // ---- P @ V --------------------------------------------------------
        Frag16 pa;
        {
            int m    = lane & 15;
            int koff = (lane >> 4) << 3;
#pragma unroll
            for (int g = 0; g < 2; ++g)
#pragma unroll
                for (int vi = 0; vi < 4; ++vi)
                    pa.u[g * 4 + vi] =
                        *(const unsigned int*)&sP[(rowb + m) * 40 + g * 16 + koff + 2 * vi];
        }
#pragma unroll
        for (int dt = 0; dt < 4; ++dt) {
            Frag16 vb;
            int dcol = dt * 16 + (lane & 15);
#pragma unroll
            for (int v = 0; v < 8; ++v)
                vb.u[v] = *(const unsigned int*)&sVT[dcol * 40 + (v >> 2) * 16 + (koffB >> 1) + 2 * (v & 3)];
            o[dt].v = __builtin_amdgcn_wmma_f32_16x16x32_bf16(
                false, pa.v, false, vb.v, (short)0, o[dt].v, false, false);
        }
    }

    // ---- write new_x (f32) ------------------------------------------------
#pragma unroll
    for (int dt = 0; dt < 4; ++dt)
#pragma unroll
        for (int r = 0; r < 8; ++r) {
            int   m   = qbase + wid * 16 + r + half8;
            float inv = 1.0f / lr[r];
            out[((size_t)(b * L + m)) * E + h * dh + dt * 16 + (lane & 15)] = o[dt].f[r] * inv;
        }
}

// ---------------------------------------------------------------------------
// LayerNorm (optionally with residual add). One block per row.
// ---------------------------------------------------------------------------
template <bool ADD>
__global__ __launch_bounds__(256) void layernorm_k(
    const float* __restrict__ a, const float* __restrict__ b2,
    const float* __restrict__ g, const float* __restrict__ beta,
    float* __restrict__ out, int n)
{
    __shared__ float red[256];
    const int    row = blockIdx.x;
    const int    t   = threadIdx.x;
    const float* pa  = a + (size_t)row * n;
    const float* pb  = ADD ? (b2 + (size_t)row * n) : nullptr;

    float sum = 0.0f, sq = 0.0f;
    for (int i = t; i < n; i += 256) {
        float v = pa[i] + (ADD ? pb[i] : 0.0f);
        sum += v; sq += v * v;
    }
    red[t] = sum; __syncthreads();
    for (int s = 128; s; s >>= 1) { if (t < s) red[t] += red[t + s]; __syncthreads(); }
    float mean = red[0] / n; __syncthreads();
    red[t] = sq; __syncthreads();
    for (int s = 128; s; s >>= 1) { if (t < s) red[t] += red[t + s]; __syncthreads(); }
    float var  = red[0] / n - mean * mean;
    float rstd = rsqrtf(var + 1e-5f);
    for (int i = t; i < n; i += 256) {
        float v = pa[i] + (ADD ? pb[i] : 0.0f);
        out[(size_t)row * n + i] = (v - mean) * rstd * g[i] + beta[i];
    }
}

// ---------------------------------------------------------------------------
// Small elementwise / precompute kernels
// ---------------------------------------------------------------------------
__global__ void sub_k(const float* a, const float* b, float* o, int n) {
    for (int i = blockIdx.x * blockDim.x + threadIdx.x; i < n; i += gridDim.x * blockDim.x)
        o[i] = a[i] - b[i];
}
__global__ void cat_k(const float* x1p, const float* x2p, unsigned short* cat, int rows, int e) {
    int n = rows * e;
    for (int i = blockIdx.x * blockDim.x + threadIdx.x; i < n; i += gridDim.x * blockDim.x) {
        int r = i / e, c = i % e;
        cat[(size_t)r * 2 * e + c]     = f2b(x1p[i]);
        cat[(size_t)r * 2 * e + e + c] = f2b(x2p[i]);
    }
}
// transpose + convert: o[c*n + r] = bf16(a[r*n + c])   (for fw -> fw^T)
__global__ void tob16t_k(const float* a, unsigned short* o, int n) {
    int nn = n * n;
    for (int i = blockIdx.x * blockDim.x + threadIdx.x; i < nn; i += gridDim.x * blockDim.x) {
        int r = i / n, c = i % n;
        o[(size_t)c * n + r] = f2b(a[i]);
    }
}
// Circ[k][j] = w[(j-k) mod n] / sqrt(n)
__global__ void circ_k(const float* w, unsigned short* o, int n, float scale) {
    int nn = n * n;
    for (int i = blockIdx.x * blockDim.x + threadIdx.x; i < nn; i += gridDim.x * blockDim.x) {
        int k = i / n, j = i % n;
        int s = j - k; if (s < 0) s += n;
        o[i] = f2b(w[s] * scale);
    }
}

// ---------------------------------------------------------------------------
// Host orchestration
// ---------------------------------------------------------------------------
extern "C" void kernel_launch(void* const* d_in, const int* in_sizes, int n_in,
                              void* d_out, int out_size, void* d_ws, size_t ws_size,
                              hipStream_t stream)
{
    (void)in_sizes; (void)n_in; (void)out_size; (void)ws_size;
    const int Bn = 8, L = 1024, E = 512, H = 8, E2 = 1024;
    const int M  = Bn * L;                       // 8192 token rows
    const size_t BLE = (size_t)M * E;            // 4,194,304

    const float* x       = (const float*)d_in[0];
    const float* attn_w  = (const float*)d_in[1];
    const float* attn_fw = (const float*)d_in[2];
    const float* attn_fb = (const float*)d_in[3];
    const float* ln0_w   = (const float*)d_in[4];
    const float* ln0_fw  = (const float*)d_in[5];
    const float* ln0_fb  = (const float*)d_in[6];
    const float* ln1_w   = (const float*)d_in[7];
    const float* ln1_fw  = (const float*)d_in[8];
    const float* ln1_fb  = (const float*)d_in[9];
    const float* ln2_w   = (const float*)d_in[10];
    const float* ln2_fw  = (const float*)d_in[11];
    const float* ln2_fb  = (const float*)d_in[12];
    const float* ln3_w   = (const float*)d_in[13];
    const float* ln3_fw  = (const float*)d_in[14];
    const float* ln3_fb  = (const float*)d_in[15];
    const float* ln4_w   = (const float*)d_in[16];
    const float* ln4_fw  = (const float*)d_in[17];
    const float* ln4_fb  = (const float*)d_in[18];
    const float* n1_g    = (const float*)d_in[19];
    const float* n1_b    = (const float*)d_in[20];
    const float* n2_g    = (const float*)d_in[21];
    const float* n2_b    = (const float*)d_in[22];

    // ---- workspace layout -------------------------------------------------
    char*  w8  = (char*)d_ws;
    size_t off = 0;
    auto take = [&](size_t bytes) -> char* {
        char* p = w8 + off;
        off += (bytes + 255) & ~(size_t)255;
        return p;
    };
    // weights stored PRE-TRANSPOSED: Wt[N][K] bf16
    unsigned short* Wt_attn = (unsigned short*)take((size_t)E * E * 2);
    unsigned short* Wt_ln0  = (unsigned short*)take((size_t)E * E * 2);
    unsigned short* Wt_ln1  = (unsigned short*)take((size_t)E * E * 2);
    unsigned short* Wt_ln2  = (unsigned short*)take((size_t)E * E * 2);
    unsigned short* Wt_ln3  = (unsigned short*)take((size_t)E2 * E2 * 2);
    unsigned short* Wt_ln4  = (unsigned short*)take((size_t)E2 * E2 * 2);
    unsigned short* circ    = (unsigned short*)take((size_t)E2 * E2 * 2);
    unsigned short* fwbT    = (unsigned short*)take((size_t)E2 * E2 * 2);
    unsigned short* qb      = (unsigned short*)take(BLE * 2);  // q in bf16
    float*          new_x   = (float*)take(BLE * 4);           // later t1
    float*          x1      = (float*)take(BLE * 4);           // later h
    float*          xln     = (float*)take(BLE * 4);
    float*          x2      = (float*)take(BLE * 4);
    unsigned short* cat     = (unsigned short*)take((size_t)M * E2 * 2);
    float*          u3      = (float*)take((size_t)M * E2 * 4);

    float* out_h   = (float*)d_out;           // layernorm(h)   [B,L,E]
    float* out_cat = (float*)d_out + BLE;     // gated output   [B,L,2E]

    // ---- precompute effective spectral weights: M_eff^T = (Circ(w) @ fw)^T
    struct Layer { const float* w; const float* fw; int n; unsigned short* Wt; };
    Layer layers[6] = {
        { attn_w, attn_fw, E,  Wt_attn }, { ln0_w, ln0_fw, E,  Wt_ln0 },
        { ln1_w,  ln1_fw,  E,  Wt_ln1  }, { ln2_w, ln2_fw, E,  Wt_ln2 },
        { ln3_w,  ln3_fw,  E2, Wt_ln3  }, { ln4_w, ln4_fw, E2, Wt_ln4 },
    };
    for (int i = 0; i < 6; ++i) {
        int n  = layers[i].n;
        int nn = n * n;
        float scale = 1.0f / sqrtf((float)n);
        tob16t_k<<<(nn + 255) / 256, 256, 0, stream>>>(layers[i].fw, fwbT, n);
        circ_k<<<(nn + 255) / 256, 256, 0, stream>>>(layers[i].w, circ, n, scale);
        gemm_wmma<false, true, true, EPI_NONE><<<dim3(n / 128, n / 64), 256, 0, stream>>>(
            circ, fwbT, nullptr, nullptr, layers[i].Wt, n, n, n);
    }

    // ---- q = x @ M_attn + b  -> bf16 (attention is its only consumer) -----
    gemm_wmma<true, true, false, EPI_BIAS><<<dim3(M / 128, E / 64), 256, 0, stream>>>(
        x, Wt_attn, attn_fb, nullptr, qb, M, E, E);

    // ---- attention --------------------------------------------------------
    attn_flash<<<dim3(Bn * H, L / 64), 128, 0, stream>>>(qb, new_x, Bn, L, E, H);

    // ---- x1 = LN(x + new_x) -----------------------------------------------
    layernorm_k<true><<<M, 256, 0, stream>>>(x, new_x, n1_g, n1_b, x1, E);

    // ---- x_ln = gelu(x1 @ M_ln0 + b0) -------------------------------------
    gemm_wmma<true, false, false, EPI_GELU><<<dim3(M / 128, E / 64), 256, 0, stream>>>(
        x1, Wt_ln0, ln0_fb, nullptr, xln, M, E, E);

    // ---- x2 = x1 - x_ln ;  cat = [new_x | x_ln] bf16 ----------------------
    sub_k<<<4096, 256, 0, stream>>>(x1, xln, x2, (int)BLE);
    cat_k<<<4096, 256, 0, stream>>>(new_x, xln, cat, M, E);

    // ---- h = sigmoid(x2@M1+b1) * (x2@M2+b2) -------------------------------
    float* t1 = new_x;   // new_x already folded into cat
    gemm_wmma<true, false, false, EPI_BIAS><<<dim3(M / 128, E / 64), 256, 0, stream>>>(
        x2, Wt_ln1, ln1_fb, nullptr, t1, M, E, E);
    float* hbuf = x1;    // x1 no longer needed
    gemm_wmma<true, false, false, EPI_GATE><<<dim3(M / 128, E / 64), 256, 0, stream>>>(
        x2, Wt_ln2, ln2_fb, t1, hbuf, M, E, E);

    // ---- output 1: LN(h) --------------------------------------------------
    layernorm_k<false><<<M, 256, 0, stream>>>(hbuf, nullptr, n2_g, n2_b, out_h, E);

    // ---- output 2: sigmoid(cat@M3+b3) * (cat@M4+b4) -----------------------
    gemm_wmma<false, false, false, EPI_BIAS><<<dim3(M / 128, E2 / 64), 256, 0, stream>>>(
        cat, Wt_ln3, ln3_fb, nullptr, u3, M, E2, E2);
    gemm_wmma<false, false, false, EPI_GATE><<<dim3(M / 128, E2 / 64), 256, 0, stream>>>(
        cat, Wt_ln4, ln4_fb, u3, out_cat, M, E2, E2);
}
